// DynamicSparseAttention_71640054497850
// MI455X (gfx1250) — matrix operand order repaired and checked by reading, not compiled, not run on hardware
//
#include <hip/hip_runtime.h>
#include <hip/hip_bf16.h>

// ---------------------------------------------------------------------------
// DynamicSparseAttention for MI455X (gfx1250, wave32, WMMA)
// B=4, S=2048, D=256, H=4, HD=64, keep top 10% per score row (q=0.9 quantile)
// ---------------------------------------------------------------------------

#define B_ 4
#define S_ 2048
#define D_ 256
#define H_ 4
#define HD_ 64

typedef __attribute__((ext_vector_type(16))) __bf16 v16bf;
typedef __attribute__((ext_vector_type(8)))  __bf16 v8bf;
typedef __attribute__((ext_vector_type(4)))  __bf16 v4bf;
typedef __attribute__((ext_vector_type(8)))  float  v8f;
typedef __attribute__((ext_vector_type(4)))  float  v4f;

// ---- CDNA5 WMMA fragment loaders (ISA 7.12.2, wave32) ----------------------
// A (16x32, 16-bit): lane loads row m = lane&15, 16 contiguous K values (this code's k order):
//   a[0..15] = A[m][16g .. 16g+15] (matches load_b_bf16_t k order)
// -> two independent 16-byte vector loads per lane.
__device__ inline v16bf load_a_bf16(const __bf16* base, int ld, int k0,
                                    int lane) {
  const int m = lane & 15, g = lane >> 4;
  const __bf16* p = base + (size_t)m * ld + k0 + 16 * g;
  v8bf lo = *(const v8bf*)p;
  v8bf hi = *(const v8bf*)(p + 8);
  v16bf a;
#pragma unroll
  for (int i = 0; i < 8; ++i) { a[i] = lo[i]; a[8 + i] = hi[i]; }
  return a;
}

// B (32x16, 16-bit): lane loads column n = lane&15, 16 consecutive K values (code's order)
// starting at k0+16g. With B stored TRANSPOSED (bt[n][k]) this is one
// contiguous 32-byte load: frag[i] = bt[n0+n][k0+16g+i].
__device__ inline v16bf load_b_bf16_t(const __bf16* bt, int ld, int k0,
                                      int n0, int lane) {
  const int n = lane & 15, g = lane >> 4;
  return *(const v16bf*)(bt + (size_t)(n0 + n) * ld + k0 + 16 * g);
}
// C/D (16x16 f32): lane 0-15 N=lane, VGPR j -> M=j; lanes 16-31 -> M=j+8.

__device__ inline unsigned f2key(float f) {
  unsigned u = __float_as_uint(f);
  return u ^ ((u >> 31) ? 0xFFFFFFFFu : 0x80000000u);
}
__device__ inline float key2f(unsigned k) {
  unsigned u = (k & 0x80000000u) ? (k ^ 0x80000000u) : ~k;
  return __uint_as_float(u);
}
__device__ inline int wave_reduce_add(int v) {
#pragma unroll
  for (int off = 16; off; off >>= 1) v += __shfl_xor(v, off, 32);
  return v;
}
__device__ inline float wave_reduce_max(float v) {
#pragma unroll
  for (int off = 16; off; off >>= 1) v = fmaxf(v, __shfl_xor(v, off, 32));
  return v;
}
__device__ inline float wave_reduce_addf(float v) {
#pragma unroll
  for (int off = 16; off; off >>= 1) v += __shfl_xor(v, off, 32);
  return v;
}

// Exact k-th smallest (0-indexed) of 2048 f32 in LDS, one wave cooperating.
__device__ inline float radix_select(const float* row, int kth, int lane) {
  unsigned prefix = 0;
  int want = kth;
#pragma unroll 1
  for (int bit = 31; bit >= 0; --bit) {
    const unsigned hiMask = (bit == 31) ? 0u : (0xFFFFFFFFu << (bit + 1));
    const unsigned bitm = 1u << bit;
    int cnt = 0;
    for (int i = lane; i < S_; i += 32) {
      const unsigned k = f2key(row[i]);
      cnt += (((k & hiMask) == prefix) && !(k & bitm)) ? 1 : 0;
    }
    cnt = wave_reduce_add(cnt);
    if (want >= cnt) { prefix |= bitm; want -= cnt; }
  }
  return key2f(prefix);
}

// ---------------------------------------------------------------------------
// Kernel 0: f32 -> bf16 pre-conversion (x and the four big weight matrices).
// One thread = 8 elements: two float4 loads -> one 16-byte bf16 store.
// ---------------------------------------------------------------------------
__global__ void __launch_bounds__(256)
cvt_kernel(const float* __restrict__ src, __bf16* __restrict__ dst) {
  const size_t i = ((size_t)blockIdx.x * 256 + threadIdx.x) * 8;
  v4f a = *(const v4f*)(src + i);
  v4f b = *(const v4f*)(src + i + 4);
  v8bf o;
#pragma unroll
  for (int j = 0; j < 4; ++j) { o[j] = (__bf16)a[j]; o[4 + j] = (__bf16)b[j]; }
  *(v8bf*)(dst + i) = o;
}

// ---------------------------------------------------------------------------
// Kernel 1: fused QKV projection. out = x @ W^T + b (all operands bf16).
// Each wave: one 16x64 tile (4 WMMA accumulators sharing the A fragment).
// Q scaled by 1/8, stored [b,h,s,hd]; K stored [b,h,s,hd] (row-major = B^T
// for QK^T); V stored transposed [b,h,hd,s] (row-major = B^T for PV).
// ---------------------------------------------------------------------------
__global__ void __launch_bounds__(256)
qkv_kernel(const __bf16* __restrict__ xb,
           const __bf16* __restrict__ Wqb, const float* __restrict__ bq,
           const __bf16* __restrict__ Wkb, const float* __restrict__ bk,
           const __bf16* __restrict__ Wvb, const float* __restrict__ bv,
           __bf16* __restrict__ q_ws, __bf16* __restrict__ k_ws,
           __bf16* __restrict__ vt_ws) {
  const int lane = threadIdx.x & 31, wave = threadIdx.x >> 5;
  const int t = blockIdx.x * 8 + wave;     // 3 * 512 * 4 = 6144 wave-tiles
  const int which = t / 2048;              // 0=Q,1=K,2=V
  const int r = t - which * 2048;
  const int m0 = (r >> 2) * 16;            // 512 M-tiles
  const int n0 = (r & 3) * 64;             // 4 N-groups of 64
  const __bf16* W   = which == 0 ? Wqb : (which == 1 ? Wkb : Wvb);
  const float* bias = which == 0 ? bq : (which == 1 ? bk : bv);

  v8f c[4] = {};
#pragma unroll 1
  for (int k0 = 0; k0 < D_; k0 += 32) {
    v16bf a = load_a_bf16(xb + (size_t)m0 * D_, D_, k0, lane);
    v16bf b0 = load_b_bf16_t(W, D_, k0, n0, lane);
    v16bf b1 = load_b_bf16_t(W, D_, k0, n0 + 16, lane);
    v16bf b2 = load_b_bf16_t(W, D_, k0, n0 + 32, lane);
    v16bf b3 = load_b_bf16_t(W, D_, k0, n0 + 48, lane);
    c[0] = __builtin_amdgcn_wmma_f32_16x16x32_bf16(false, a, false, b0,
                                                   (short)0, c[0], false, false);
    c[1] = __builtin_amdgcn_wmma_f32_16x16x32_bf16(false, a, false, b1,
                                                   (short)0, c[1], false, false);
    c[2] = __builtin_amdgcn_wmma_f32_16x16x32_bf16(false, a, false, b2,
                                                   (short)0, c[2], false, false);
    c[3] = __builtin_amdgcn_wmma_f32_16x16x32_bf16(false, a, false, b3,
                                                   (short)0, c[3], false, false);
  }

  const int g = lane >> 4, n = lane & 15;
#pragma unroll
  for (int tt = 0; tt < 4; ++tt) {
#pragma unroll
    for (int j = 0; j < 8; ++j) {
      const int mrow = m0 + j + 8 * g;
      const int col = n0 + tt * 16 + n;
      float val = c[tt][j] + bias[col];
      const int b = mrow >> 11, s = mrow & (S_ - 1);
      const int h = col >> 6, hd = col & 63;
      const int bh = b * H_ + h;
      if (which == 0) {
        q_ws[((size_t)bh * S_ + s) * HD_ + hd] = (__bf16)(val * 0.125f);
      } else if (which == 1) {
        k_ws[((size_t)bh * S_ + s) * HD_ + hd] = (__bf16)val;
      } else {
        vt_ws[((size_t)bh * HD_ + hd) * S_ + s] = (__bf16)val;
      }
    }
  }
}

// ---------------------------------------------------------------------------
// Kernel 2: fused attention for one (b,h,16-query tile).
// LDS: scores 16x2048 f32 (128KB) + probs 16x2048 bf16 (64KB) = 192KB < 320KB.
// ---------------------------------------------------------------------------
__global__ void __launch_bounds__(256)
attn_kernel(const __bf16* __restrict__ q_ws, const __bf16* __restrict__ k_ws,
            const __bf16* __restrict__ vt_ws, float* __restrict__ att_ws) {
  extern __shared__ char smem[];
  float*  sc = (float*)smem;                       // [16][2048]
  __bf16* pb = (__bf16*)(smem + 16 * S_ * 4);      // [16][2048]

  const int lane = threadIdx.x & 31, wave = threadIdx.x >> 5;
  const int bh = blockIdx.x >> 7;
  const int q0 = (blockIdx.x & 127) * 16;
  const int b = bh >> 2, h = bh & 3;

  // ---- Phase 1: scores = (Q/8) K^T. Hoist the two Q A-fragments. ----
  const __bf16* qbase = q_ws + ((size_t)bh * S_ + q0) * HD_;
  v16bf qa0 = load_a_bf16(qbase, HD_, 0, lane);
  v16bf qa1 = load_a_bf16(qbase, HD_, 32, lane);

  const __bf16* kb = k_ws + (size_t)bh * S_ * HD_;  // rows of K = rows of B^T
  const int g = lane >> 4, nn = lane & 15;
#pragma unroll 1
  for (int tt = 0; tt < 16; ++tt) {                // 16 key tiles per wave
    const int key0 = (wave * 16 + tt) * 16;
    v16bf b0 = load_b_bf16_t(kb, HD_, 0, key0, lane);
    v16bf b1 = load_b_bf16_t(kb, HD_, 32, key0, lane);
    v8f c = {};
    c = __builtin_amdgcn_wmma_f32_16x16x32_bf16(false, qa0, false, b0,
                                                (short)0, c, false, false);
    c = __builtin_amdgcn_wmma_f32_16x16x32_bf16(false, qa1, false, b1,
                                                (short)0, c, false, false);
#pragma unroll
    for (int j = 0; j < 8; ++j)
      sc[(size_t)(j + 8 * g) * S_ + key0 + nn] = c[j];
  }
  __syncthreads();

  // ---- Phase 2: exact quantile + masked softmax (2 rows per wave) ----
  const float pos = 0.9f * (float)(S_ - 1);        // 1842.3
  const int   lo = (int)pos;
  const float frac = pos - (float)lo;
#pragma unroll 1
  for (int rr = 0; rr < 2; ++rr) {
    const int r = wave * 2 + rr;
    float* row = sc + (size_t)r * S_;
    const float v0 = radix_select(row, lo, lane);
    const float v1 = radix_select(row, lo + 1, lane);
    const float thr = v0 + frac * (v1 - v0);

    float mx = -3.402823466e38f;
    for (int i = lane; i < S_; i += 32) mx = fmaxf(mx, row[i]);
    mx = wave_reduce_max(mx);                      // row max >= thr always

    float sum = 0.f;
    for (int i = lane; i < S_; i += 32) {
      const float xv = row[i];
      sum += (xv >= thr) ? __expf(xv - mx) : 0.f;
    }
    sum = wave_reduce_addf(sum);
    const float inv = 1.f / sum;
    for (int i = lane; i < S_; i += 32) {
      const float xv = row[i];
      const float p = (xv >= thr) ? __expf(xv - mx) * inv : 0.f;
      pb[(size_t)r * S_ + i] = (__bf16)p;
    }
  }
  __syncthreads();

  // ---- Phase 3: att = probs @ V. 8 waves: 4 N-tiles x 2 K-halves. ----
  float* part = sc;                                 // reuse: [8][16][16]
  const int ntile = wave & 3, khalf = wave >> 2;
  const int n0 = ntile * 16, kstart = khalf * (S_ / 2);
  const __bf16* vtb = vt_ws + (size_t)bh * HD_ * S_;  // V^T rows = B^T rows
  v8f c = {};
#pragma unroll 1
  for (int k0 = kstart; k0 < kstart + S_ / 2; k0 += 32) {
    __builtin_prefetch(vtb + (size_t)n0 * S_ + k0 + 256, 0, 3);
    v16bf a = load_a_bf16(pb, S_, k0, lane);        // ds_load 16B chunks
    v16bf bf = load_b_bf16_t(vtb, S_, k0, n0, lane);
    c = __builtin_amdgcn_wmma_f32_16x16x32_bf16(false, a, false, bf,
                                                (short)0, c, false, false);
  }
#pragma unroll
  for (int j = 0; j < 8; ++j)
    part[(size_t)wave * 256 + (j + 8 * g) * 16 + nn] = c[j];
  __syncthreads();

  // Combine K-halves, store att in [b,s, h*64+hd] f32 layout.
  for (int idx = threadIdx.x; idx < 16 * HD_; idx += 256) {
    const int m = idx >> 6, d = idx & 63;
    const int t = d >> 4, n = d & 15;
    const float val = part[(size_t)t * 256 + m * 16 + n] +
                      part[(size_t)(t + 4) * 256 + m * 16 + n];
    att_ws[((size_t)(b * S_ + q0 + m)) * D_ + h * HD_ + d] = val;
  }
}

// ---------------------------------------------------------------------------
// Kernel 3: gate per batch: g = sigmoid(mean_s(x) @ Wg^T + bg) @ Wgp^T + bgp
// ---------------------------------------------------------------------------
__global__ void __launch_bounds__(256)
gate_kernel(const float* __restrict__ x, const float* __restrict__ Wg,
            const float* __restrict__ bg, const float* __restrict__ Wgp,
            const float* __restrict__ bgp, float* __restrict__ gp_ws) {
  __shared__ float xm[D_];
  __shared__ float gls[16];
  const int b = blockIdx.x, d = threadIdx.x;
  const float* xb = x + (size_t)b * S_ * D_;
  float acc = 0.f;
  for (int s = 0; s < S_; ++s) acc += xb[(size_t)s * D_ + d];
  xm[d] = acc * (1.0f / (float)S_);
  __syncthreads();
  if (d < 16) {
    float gv = bg[d];
    for (int i = 0; i < D_; ++i) gv += xm[i] * Wg[d * D_ + i];
    gls[d] = 1.f / (1.f + __expf(-gv));
  }
  __syncthreads();
  float gp = bgp[d];
#pragma unroll
  for (int j = 0; j < 16; ++j) gp += gls[j] * Wgp[d * 16 + j];
  gp_ws[b * D_ + d] = gp;
}

// ---------------------------------------------------------------------------
// Kernel 4: distill (16x16) + gated mix -> bf16 mixed
// ---------------------------------------------------------------------------
__global__ void __launch_bounds__(256)
mix_kernel(const float* __restrict__ att_ws, const float* __restrict__ Wd,
           const float* __restrict__ bd, const float* __restrict__ gp_ws,
           __bf16* __restrict__ mixed) {
  __shared__ float rowbuf[D_];
  __shared__ float wd[256];
  __shared__ float bdl[16];
  const int row = blockIdx.x, e = threadIdx.x;
  const int b = row >> 11;
  rowbuf[e] = att_ws[(size_t)row * D_ + e];
  wd[e] = Wd[e];
  if (e < 16) bdl[e] = bd[e];
  __syncthreads();
  const int chunk = e >> 4, j = e & 15;
  float dv = bdl[j];
#pragma unroll
  for (int i = 0; i < 16; ++i) dv += rowbuf[chunk * 16 + i] * wd[j * 16 + i];
  const float g = gp_ws[b * D_ + e];
  const float m = g * dv + (1.f - g) * rowbuf[e];
  mixed[(size_t)row * D_ + e] = (__bf16)m;
}

// ---------------------------------------------------------------------------
// Kernel 5: output projection. out = mixed @ Wo^T + bo (f32 out).
// 16x64 tile per wave, 4 accumulators sharing each A fragment.
// ---------------------------------------------------------------------------
__global__ void __launch_bounds__(256)
out_gemm_kernel(const __bf16* __restrict__ mixed,
                const __bf16* __restrict__ Wob,
                const float* __restrict__ bo, float* __restrict__ out) {
  const int lane = threadIdx.x & 31, wave = threadIdx.x >> 5;
  const int t = blockIdx.x * 8 + wave;   // 512 * 4 = 2048 wave-tiles
  const int m0 = (t >> 2) * 16;
  const int n0 = (t & 3) * 64;
  v8f c[4] = {};
#pragma unroll 1
  for (int k0 = 0; k0 < D_; k0 += 32) {
    v16bf a = load_a_bf16(mixed + (size_t)m0 * D_, D_, k0, lane);
    v16bf b0 = load_b_bf16_t(Wob, D_, k0, n0, lane);
    v16bf b1 = load_b_bf16_t(Wob, D_, k0, n0 + 16, lane);
    v16bf b2 = load_b_bf16_t(Wob, D_, k0, n0 + 32, lane);
    v16bf b3 = load_b_bf16_t(Wob, D_, k0, n0 + 48, lane);
    c[0] = __builtin_amdgcn_wmma_f32_16x16x32_bf16(false, a, false, b0,
                                                   (short)0, c[0], false, false);
    c[1] = __builtin_amdgcn_wmma_f32_16x16x32_bf16(false, a, false, b1,
                                                   (short)0, c[1], false, false);
    c[2] = __builtin_amdgcn_wmma_f32_16x16x32_bf16(false, a, false, b2,
                                                   (short)0, c[2], false, false);
    c[3] = __builtin_amdgcn_wmma_f32_16x16x32_bf16(false, a, false, b3,
                                                   (short)0, c[3], false, false);
  }
  const int g = lane >> 4, n = lane & 15;
#pragma unroll
  for (int tt = 0; tt < 4; ++tt) {
#pragma unroll
    for (int j = 0; j < 8; ++j) {
      const int mrow = m0 + j + 8 * g;
      const int col = n0 + tt * 16 + n;
      out[(size_t)mrow * D_ + col] = c[tt][j] + bo[col];
    }
  }
}

// ---------------------------------------------------------------------------
extern "C" void kernel_launch(void* const* d_in, const int* in_sizes, int n_in,
                              void* d_out, int out_size, void* d_ws,
                              size_t ws_size, hipStream_t stream) {
  const float* x   = (const float*)d_in[0];
  const float* Wq  = (const float*)d_in[1];
  const float* bq  = (const float*)d_in[2];
  const float* Wk  = (const float*)d_in[3];
  const float* bk  = (const float*)d_in[4];
  const float* Wv  = (const float*)d_in[5];
  const float* bv  = (const float*)d_in[6];
  const float* Wd  = (const float*)d_in[7];
  const float* bd  = (const float*)d_in[8];
  const float* Wg  = (const float*)d_in[9];
  const float* bg  = (const float*)d_in[10];
  const float* Wgp = (const float*)d_in[11];
  const float* bgp = (const float*)d_in[12];
  const float* Wo  = (const float*)d_in[13];
  const float* bo  = (const float*)d_in[14];
  float* out = (float*)d_out;

  // Workspace carve-up (~28.5 MB total)
  char* ws = (char*)d_ws;
  const size_t nElem = (size_t)B_ * S_ * D_;       // 2M elements
  const size_t wElem = (size_t)D_ * D_;            // 64K elements
  size_t o = 0;
  __bf16* q_ws  = (__bf16*)(ws + o); o += nElem * 2;   // 4 MB
  __bf16* k_ws  = (__bf16*)(ws + o); o += nElem * 2;   // 4 MB
  __bf16* vt_ws = (__bf16*)(ws + o); o += nElem * 2;   // 4 MB
  float*  att_ws = (float*)(ws + o); o += nElem * 4;   // 8 MB
  __bf16* mixed  = (__bf16*)(ws + o); o += nElem * 2;  // 4 MB
  __bf16* xb   = (__bf16*)(ws + o); o += nElem * 2;    // 4 MB
  __bf16* Wqb  = (__bf16*)(ws + o); o += wElem * 2;    // 128 KB
  __bf16* Wkb  = (__bf16*)(ws + o); o += wElem * 2;
  __bf16* Wvb  = (__bf16*)(ws + o); o += wElem * 2;
  __bf16* Wob  = (__bf16*)(ws + o); o += wElem * 2;
  float*  gp_ws = (float*)(ws + o); o += (size_t)B_ * D_ * 4;  // 4 KB

  // 0) Pre-convert f32 -> bf16 (8 elems/thread)
  cvt_kernel<<<(int)(nElem / (256 * 8)), 256, 0, stream>>>(x, xb);
  cvt_kernel<<<(int)(wElem / (256 * 8)), 256, 0, stream>>>(Wq, Wqb);
  cvt_kernel<<<(int)(wElem / (256 * 8)), 256, 0, stream>>>(Wk, Wkb);
  cvt_kernel<<<(int)(wElem / (256 * 8)), 256, 0, stream>>>(Wv, Wvb);
  cvt_kernel<<<(int)(wElem / (256 * 8)), 256, 0, stream>>>(Wo, Wob);

  // 1) QKV projections: 6144 wave-tiles / 8 waves = 768 blocks
  qkv_kernel<<<768, 256, 0, stream>>>(xb, Wqb, bq, Wkb, bk, Wvb, bv,
                                      q_ws, k_ws, vt_ws);

  // 2) Fused sparse attention: B*H*(S/16) = 2048 blocks, 192 KB dynamic LDS
  const int attn_lds = 16 * S_ * 4 + 16 * S_ * 2;  // 196608 B < 320 KB WGP LDS
  attn_kernel<<<2048, 256, attn_lds, stream>>>(q_ws, k_ws, vt_ws, att_ws);

  // 3) Gate per batch
  gate_kernel<<<B_, 256, 0, stream>>>(x, Wg, bg, Wgp, bgp, gp_ws);

  // 4) Distill + mix
  mix_kernel<<<B_ * S_, 256, 0, stream>>>(att_ws, Wd, bd, gp_ws, mixed);

  // 5) Output projection: 2048 wave-tiles / 8 waves = 256 blocks
  out_gemm_kernel<<<256, 256, 0, stream>>>(mixed, Wob, bo, out);
}
